// TTRCell_15779709846166
// MI455X (gfx1250) — matrix operand order repaired and checked by reading, not compiled, not run on hardware
//
#include <hip/hip_runtime.h>

// ---------------------------------------------------------------------------
// Chunked fast-weight linear attention for MI455X (gfx1250, wave32, WMMA).
// q,k: [BH,S,F] f32; v: [BH,S,V] f32; out: [BH,S,V] f32. BH=32,S=4096,F=V=128.
//   1) S_c = K_c^T V_c per (bh,chunk)            [1024 blocks, WMMA bf16]
//   2) inclusive prefix of S_c over chunks       [elementwise, coalesced]
//   3) out_c = mask(Q K^T) V + Q W_{c-1}, scaled [1024 blocks, WMMA bf16]
// All WMMA operand fragments are contiguous 16B runs in LDS -> ds_load_b128.
// Workspace: BH*NC*F*V f32 = 64 MiB in d_ws.
// ---------------------------------------------------------------------------

typedef __attribute__((ext_vector_type(16))) __bf16 v16bf;
typedef __attribute__((ext_vector_type(8)))  __bf16 v8bf;
typedef __attribute__((ext_vector_type(4)))  __bf16 v4bf;
typedef __attribute__((ext_vector_type(8)))  float  v8f;

#define BH   32
#define SEQ  4096
#define FD   128
#define VDIM 128
#define CK   128
#define NC   (SEQ / CK)
#define LDP  136   // LDS row stride (bf16): 272B, 16B-aligned, conflict-breaking

__device__ __forceinline__ __bf16 f2bf(float x) { return (__bf16)x; }

// --- fragment loads (CDNA5 ISA 7.12.2 layouts, wave32) ----------------------
// A operand (16x32 MxK) from row-major [m][k] LDS: lane row m=lane&15,
// elements j=0..7 at k = kb + 8*(lane>>4) + j, j=8..15 at +16. Two b128 loads.
__device__ __forceinline__ v16bf load_fragA(const __bf16* p, int mb, int kb, int lane) {
  const __bf16* r = p + (mb + (lane & 15)) * LDP + kb + ((lane >> 4) << 3);
  const v8bf* pl = (const v8bf*)__builtin_assume_aligned(r, 16);
  const v8bf* ph = (const v8bf*)__builtin_assume_aligned(r + 16, 16);
  v8bf lo = *pl, hi = *ph;
  return __builtin_shufflevector(lo, hi, 0,1,2,3,4,5,6,7,8,9,10,11,12,13,14,15);
}
// B operand (32x16 KxN) from a ROW-MAJOR [n][k] LDS image (i.e. K^T stored
// row-major, or a transposed-staged matrix): lane row n=lane&15, elements
// j=0..7 at k = kb + 8*(lane>>4) + j, j=8..15 at +16. Two b128 loads.
__device__ __forceinline__ v16bf load_fragB(const __bf16* p, int nb, int kb, int lane) {
  const __bf16* r = p + (nb + (lane & 15)) * LDP + kb + ((lane >> 4) << 3);
  const v8bf* pl = (const v8bf*)__builtin_assume_aligned(r, 16);
  const v8bf* ph = (const v8bf*)__builtin_assume_aligned(r + 16, 16);
  v8bf lo = *pl, hi = *ph;
  return __builtin_shufflevector(lo, hi, 0,1,2,3,4,5,6,7,8,9,10,11,12,13,14,15);
}

// --- staging: 128x128 f32 row-major global -> bf16 LDS ----------------------
// Row-major copy: vector read, one 8B vector store.
__device__ __forceinline__ void stage_rm(__bf16* dst, const float* __restrict__ src, int tid) {
#pragma unroll
  for (int it = 0; it < 16; ++it) {
    int i  = tid + it * 256;
    int r  = i >> 5;
    int c4 = (i & 31) << 2;
    float4 f = reinterpret_cast<const float4*>(src)[i];
    v4bf w = { f2bf(f.x), f2bf(f.y), f2bf(f.z), f2bf(f.w) };
    *(v4bf*)__builtin_assume_aligned(dst + r * LDP + c4, 8) = w;
  }
}
// Transposed copy: dst[c][r] = src[r][c]; 4 scattered b16 stores per float4.
__device__ __forceinline__ void stage_tr(__bf16* dst, const float* __restrict__ src, int tid) {
#pragma unroll
  for (int it = 0; it < 16; ++it) {
    int i  = tid + it * 256;
    int r  = i >> 5;
    int c4 = (i & 31) << 2;
    float4 f = reinterpret_cast<const float4*>(src)[i];
    dst[(c4 + 0) * LDP + r] = f2bf(f.x);
    dst[(c4 + 1) * LDP + r] = f2bf(f.y);
    dst[(c4 + 2) * LDP + r] = f2bf(f.z);
    dst[(c4 + 3) * LDP + r] = f2bf(f.w);
  }
}

#define WMMA_BF16(a, b, c) \
  __builtin_amdgcn_wmma_f32_16x16x32_bf16(false, (a), false, (b), (short)0, (c), false, false)

// --------------------------- Pass 1: S_c = K^T V ----------------------------
__global__ __launch_bounds__(256) void kv_chunk_kernel(const float* __restrict__ K,
                                                       const float* __restrict__ V,
                                                       float* __restrict__ Sws) {
  __shared__ __align__(16) __bf16 ktr[FD * LDP];    // ktr[f][s] = K[s][f]
  __shared__ __align__(16) __bf16 vtr[VDIM * LDP];  // vtr[v][s] = V[s][v]
  const int c = blockIdx.x, bh = blockIdx.y, tid = threadIdx.x;
  const size_t base = ((size_t)bh * SEQ + (size_t)c * CK) * FD;
  stage_tr(ktr, K + base, tid);
  stage_tr(vtr, V + base, tid);
  __syncthreads();

  const int lane = tid & 31, wave = tid >> 5;
  const int m_base = wave * 16;                     // F rows owned by this wave
  float* outp = Sws + ((size_t)bh * NC + c) * (size_t)FD * VDIM;

  v16bf afrag[4];                                   // A[f][s] = ktr row-major
#pragma unroll
  for (int kt = 0; kt < 4; ++kt) afrag[kt] = load_fragA(ktr, m_base, kt * 32, lane);

#pragma unroll
  for (int nt = 0; nt < 8; ++nt) {
    v8f acc = {};
#pragma unroll
    for (int kt = 0; kt < 4; ++kt) {
      v16bf b = load_fragB(vtr, nt * 16, kt * 32, lane);   // B[s][v] via vtr[v][s]
      acc = WMMA_BF16(afrag[kt], b, acc);
    }
    const int n    = nt * 16 + (lane & 15);
    const int mrow = m_base + ((lane >> 4) << 3);
#pragma unroll
    for (int r = 0; r < 8; ++r)
      outp[(size_t)(mrow + r) * VDIM + n] = acc[r];
  }
}

// ------------------- Pass 2: inclusive prefix over chunks -------------------
__global__ __launch_bounds__(128) void prefix_kernel(float* __restrict__ Sws) {
  const int f = blockIdx.x, bh = blockIdx.y, v = threadIdx.x;
  float acc = 0.0f;
  size_t idx = (((size_t)bh * NC) * FD + f) * VDIM + v;
  const size_t step = (size_t)FD * VDIM;
#pragma unroll 1
  for (int c = 0; c < NC; ++c, idx += step) {
    acc += Sws[idx];
    Sws[idx] = acc;
  }
}

// ----------- Pass 3: out = mask(Q K^T) V + Q W_{c-1}, scaled ----------------
__global__ __launch_bounds__(256) void out_kernel(const float* __restrict__ Q,
                                                  const float* __restrict__ K,
                                                  const float* __restrict__ V,
                                                  const float* __restrict__ Sws,
                                                  float* __restrict__ Out) {
  __shared__ __align__(16) __bf16 qs [CK * LDP];    // Q row-major [m][f]
  __shared__ __align__(16) __bf16 ksA[CK * LDP];    // K row-major [n][f]; then masked A [m][n]
  __shared__ __align__(16) __bf16 vtr[VDIM * LDP];  // vtr[v][s] = V[s][v]
  __shared__ __align__(16) __bf16 wtr[VDIM * LDP];  // wtr[v][f] = W_prev[f][v]
  const int c = blockIdx.x, bh = blockIdx.y, tid = threadIdx.x;
  const size_t base = ((size_t)bh * SEQ + (size_t)c * CK) * FD;
  stage_rm(qs, Q + base, tid);
  stage_rm(ksA, K + base, tid);
  stage_tr(vtr, V + base, tid);
  if (c > 0) {
    stage_tr(wtr, Sws + ((size_t)bh * NC + (c - 1)) * (size_t)FD * VDIM, tid);
  } else {
    for (int i = tid; i < VDIM * LDP; i += 256) wtr[i] = f2bf(0.0f);
  }
  __syncthreads();

  const int lane = tid & 31, wave = tid >> 5;
  const int m_base = wave * 16;

  v16bf qfrag[4];                                   // reused in both phases
#pragma unroll
  for (int kt = 0; kt < 4; ++kt) qfrag[kt] = load_fragA(qs, m_base, kt * 32, lane);

  // Phase 1: A = Q K^T  (B operand = K rows read row-major = K^T fragments)
  v8f at[8];
#pragma unroll
  for (int nt = 0; nt < 8; ++nt) {
    v8f acc = {};
#pragma unroll
    for (int kt = 0; kt < 4; ++kt) {
      v16bf b = load_fragB(ksA, nt * 16, kt * 32, lane);   // B[f][n] = K[n][f]
      acc = WMMA_BF16(qfrag[kt], b, acc);
    }
    at[nt] = acc;
  }
  __syncthreads();   // all waves done reading K before it is overwritten

  // Causal mask (inclusive) -> bf16 A over the K buffer, row-major [m][n]
  {
    const int ncol = lane & 15;
    const int mrow = m_base + ((lane >> 4) << 3);
#pragma unroll
    for (int nt = 0; nt < 8; ++nt) {
      const int n = nt * 16 + ncol;
#pragma unroll
      for (int r = 0; r < 8; ++r) {
        const int m = mrow + r;
        float x = (n <= m) ? at[nt][r] : 0.0f;
        ksA[m * LDP + n] = f2bf(x);
      }
    }
  }
  __syncthreads();

  // Phase 2: out = A V + Q W_prev, scaled by 1/(global_t + 1)
  v16bf afrag[4];
#pragma unroll
  for (int kt = 0; kt < 4; ++kt) afrag[kt] = load_fragA(ksA, m_base, kt * 32, lane);

#pragma unroll
  for (int nt = 0; nt < 8; ++nt) {
    v8f acc = {};
#pragma unroll
    for (int kt = 0; kt < 4; ++kt) {
      v16bf bv = load_fragB(vtr, nt * 16, kt * 32, lane);  // B[s][v] via vtr[v][s]
      acc = WMMA_BF16(afrag[kt], bv, acc);
      v16bf bw = load_fragB(wtr, nt * 16, kt * 32, lane);  // B[f][v] via wtr[v][f]
      acc = WMMA_BF16(qfrag[kt], bw, acc);
    }
    const int n    = nt * 16 + (lane & 15);
    const int mrow = m_base + ((lane >> 4) << 3);
#pragma unroll
    for (int r = 0; r < 8; ++r) {
      const int m = mrow + r;
      const float sc = 1.0f / (float)(c * CK + m + 1);
      Out[((size_t)bh * SEQ + (size_t)c * CK + m) * VDIM + n] = acc[r] * sc;
    }
  }
}

// ---------------------------------------------------------------------------
extern "C" void kernel_launch(void* const* d_in, const int* in_sizes, int n_in,
                              void* d_out, int out_size, void* d_ws, size_t ws_size,
                              hipStream_t stream) {
  const float* q = (const float*)d_in[0];
  const float* k = (const float*)d_in[1];
  const float* v = (const float*)d_in[2];
  float* out = (float*)d_out;
  float* sws = (float*)d_ws;   // requires BH*NC*FD*VDIM*4 = 64 MiB of scratch

  dim3 gridc(NC, BH);          // 32 x 32 = 1024 blocks
  kv_chunk_kernel<<<gridc, 256, 0, stream>>>(k, v, sws);
  prefix_kernel<<<dim3(FD, BH), VDIM, 0, stream>>>(sws);
  out_kernel<<<gridc, 256, 0, stream>>>(q, k, v, sws, out);

  (void)in_sizes; (void)n_in; (void)out_size; (void)ws_size;
}
